// DecoderBlock_30434138259804
// MI455X (gfx1250) — matrix operand order repaired and checked by reading, not compiled, not run on hardware
//
#include <hip/hip_runtime.h>
#include <hip/hip_bf16.h>
#include <math.h>

// ---------------------------------------------------------------------------
// Types for WMMA fragments (gfx1250: v_wmma_f32_16x16x32_bf16)
// ---------------------------------------------------------------------------
typedef __bf16 bf16;
typedef __bf16 bf16x8  __attribute__((ext_vector_type(8)));
typedef __bf16 bf16x16 __attribute__((ext_vector_type(16)));
typedef float  floatx8 __attribute__((ext_vector_type(8)));
typedef unsigned int uint32x4 __attribute__((ext_vector_type(4)));
typedef int int32x4 __attribute__((ext_vector_type(4)));
typedef int int32x8 __attribute__((ext_vector_type(8)));

// A-fragment: lane holds row m=lane%16; k = half*8..+7 and half*8+16..+23.
static __device__ inline bf16x16 frag_a(const bf16* p) {
    bf16x8 lo = *(const bf16x8*)p;
    bf16x8 hi = *(const bf16x8*)(p + 16);
    return __builtin_shufflevector(lo, hi, 0,1,2,3,4,5,6,7,8,9,10,11,12,13,14,15);
}
// B-fragment: lane holds col n=lane%16; k = half*8..+7 and half*8+16..+23.
static __device__ inline bf16x16 frag_b(const bf16* p) {
    bf16x8 lo = *(const bf16x8*)p;
    bf16x8 hi = *(const bf16x8*)(p + 16);
    return __builtin_shufflevector(lo, hi, 0,1,2,3,4,5,6,7,8,9,10,11,12,13,14,15);
}
#define WMMA_BF16(A, B, C) \
    __builtin_amdgcn_wmma_f32_16x16x32_bf16(false, (A), false, (B), (short)0, (C), false, false)

// Low 32 bits of a generic pointer to LDS == LDS byte address (aperture rule).
#define LDS_U32(p) ((unsigned)(size_t)(p))

// CDNA5 async global->LDS copy (ASYNCcnt-tracked), 16B per lane.
static __device__ inline void async_copy_b128(unsigned lds, unsigned long long gaddr) {
    asm volatile("global_load_async_to_lds_b128 %0, %1, off"
                 :: "v"(lds), "v"(gaddr) : "memory");
}
static __device__ inline void wait_async0() {
    asm volatile("s_wait_asynccnt 0x0" ::: "memory");
}

// CDNA5 Tensor Data Mover: 2D bf16 tile load, tileW=64 elems (128B rows),
// LDS pad 4 dwords (16B) after every 32 dwords (128B) -> 72-elem LDS rows.
static __device__ inline void tdm_load_tile64(unsigned lds_addr, const void* gptr,
                                              unsigned strideElems) {
    unsigned long long ga = (unsigned long long)gptr;
    uint32x4 g0;
    g0[0] = 1u;                                   // count=1 valid descriptor
    g0[1] = lds_addr;                             // lds_addr [63:32]
    g0[2] = (unsigned)ga;                         // global_addr [95:64]
    g0[3] = ((unsigned)(ga >> 32) & 0x1FFFFFFu)   // global_addr [120:96]
          | 0x80000000u;                          // type=2 ("image") [127:126]
    int32x8 g1;
    g1[0] = (int)((1u << 16)      // data_size = 1 -> 2 bytes
                | (1u << 20)      // pad_enable
                | (4u << 22)      // pad_interval: code 4 = 32 dwords (128B)
                | (3u << 25));    // pad_amount:  code 3 = 4 dwords (16B)
    g1[1] = (int)(64u << 16);                     // tensor_dim0 = 64 (lo16)
    g1[2] = (int)(64u << 16);                     // dim0 hi=0 | tensor_dim1 = 64 (lo16)
    g1[3] = (int)(64u << 16);                     // dim1 hi=0 | tile_dim0 = 64
    g1[4] = (int)64;                              // tile_dim1 = 64, tile_dim2 = 0
    g1[5] = (int)strideElems;                     // tensor_dim0_stride lo32
    g1[6] = 0;
    g1[7] = 0;
    int32x4 z4 = {0, 0, 0, 0};
    int32x8 z8 = {0, 0, 0, 0, 0, 0, 0, 0};
    __builtin_amdgcn_tensor_load_to_lds(g0, g1, z4, z4, z8, 0);
}

// ---------------------------------------------------------------------------
// Problem constants
// ---------------------------------------------------------------------------
#define Bsz  2
#define Lsq  2048
#define Dm   768
#define Hn   12
#define FFd  3072
#define DHd  64
#define ROWS (Bsz * Lsq)           // 4096 tokens
#define MODC (6 * Dm)              // 4608
#define SQRT_D 27.712812921102035f // sqrt(768)

// ---------------------------------------------------------------------------
// One-time: W[K,N] fp32 -> Wt[N,K] bf16 (tiled transpose + convert)
// ---------------------------------------------------------------------------
__global__ __launch_bounds__(256) void transpose_convert_kernel(
    const float* __restrict__ W, bf16* __restrict__ Wt, int K, int N)
{
    __shared__ float t[32][33];
    const int k0 = blockIdx.y * 32, n0 = blockIdx.x * 32;
    const int c = threadIdx.x & 31, r8 = threadIdx.x >> 5;
#pragma unroll
    for (int i = 0; i < 4; ++i)
        t[r8 + i * 8][c] = W[(size_t)(k0 + r8 + i * 8) * N + n0 + c];
    __syncthreads();
#pragma unroll
    for (int i = 0; i < 4; ++i)
        Wt[(size_t)(n0 + r8 + i * 8) * K + k0 + c] = (bf16)t[c][r8 + i * 8];
}

// silu(x) fp32 -> bf16 elementwise
__global__ __launch_bounds__(256) void silu_bf16_kernel(
    const float* __restrict__ in, bf16* __restrict__ out)
{
    size_t i = (size_t)blockIdx.x * 256 + threadIdx.x;
    float v = in[i];
    out[i] = (bf16)(v / (1.0f + __expf(-v)));
}

// ---------------------------------------------------------------------------
// bf16 WMMA GEMM with async double-buffered LDS staging.
// A[M,K] bf16 row-major, Wt[N,K] bf16 row-major (pre-transposed weights).
// EPI=0: C fp32 = acc + bias.  EPI=1: C bf16 = silu(acc + bias).
// Block 256 (8 waves): tile 128x128, wave = 32x64, BK=32.
// ---------------------------------------------------------------------------
#define TILEB (128 * 48 * 2) // bytes per LDS tile buffer

template <int EPI>
__global__ __launch_bounds__(256) void gemm_bf16_kernel(
    const bf16* __restrict__ A, const bf16* __restrict__ Wt,
    const float* __restrict__ bias, void* __restrict__ Cout,
    int M, int N, int K)
{
    const int tid  = threadIdx.x;
    const int lane = tid & 31;
    const int wv   = tid >> 5;
    const int ln   = lane & 15;
    const int half = lane >> 4;
    const int row0 = blockIdx.y * 128;
    const int col0 = blockIdx.x * 128;
    const int wm   = (wv & 3) * 32;
    const int wn   = (wv >> 2) * 64;

    __shared__ __align__(16) bf16 As[2][128 * 48];
    __shared__ __align__(16) bf16 Bs[2][128 * 48];
    const unsigned asBase = LDS_U32(&As[0][0]);
    const unsigned bsBase = LDS_U32(&Bs[0][0]);

    floatx8 acc[2][4] = {};
    const int T = K / 32;

    // Stage 256x32 of A + 256x32 of Wt: 2048 16B chunks, 8/thread.
    auto issue = [&](int t, int buf) {
#pragma unroll
        for (int p = 0; p < 2; ++p) {
#pragma unroll
            for (int j = 0; j < 2; ++j) {
                int chunk = p * 512 + 2 * tid + j;
                int rr = chunk >> 2;      // tile row 0..255
                int cc = chunk & 3;       // 16B chunk within 32-elem k-row
                async_copy_b128(asBase + buf * TILEB + rr * 96 + cc * 16,
                    (unsigned long long)(A + (size_t)(row0 + rr) * K + t * 32 + cc * 8));
                async_copy_b128(bsBase + buf * TILEB + rr * 96 + cc * 16,
                    (unsigned long long)(Wt + (size_t)(col0 + rr) * K + t * 32 + cc * 8));
            }
        }
    };

    issue(0, 0);
    for (int t = 0; t < T; ++t) {
        wait_async0();
        __syncthreads();
        if (t + 1 < T) issue(t + 1, (t + 1) & 1);
        const bf16* as = &As[t & 1][0];
        const bf16* bs = &Bs[t & 1][0];
        bf16x16 af[2], bfg[4];
#pragma unroll
        for (int i = 0; i < 2; ++i)
            af[i] = frag_a(&as[(wm + i * 16 + ln) * 48 + half * 8]);
#pragma unroll
        for (int j = 0; j < 4; ++j)
            bfg[j] = frag_b(&bs[(wn + j * 16 + ln) * 48 + half * 8]);
#pragma unroll
        for (int i = 0; i < 2; ++i)
#pragma unroll
            for (int j = 0; j < 4; ++j)
                acc[i][j] = WMMA_BF16(af[i], bfg[j], acc[i][j]);
        __syncthreads();
    }

#pragma unroll
    for (int i = 0; i < 2; ++i)
#pragma unroll
        for (int j = 0; j < 4; ++j)
#pragma unroll
            for (int r = 0; r < 8; ++r) {
                int m = row0 + wm + i * 16 + r + half * 8;
                int n = col0 + wn + j * 16 + ln;
                float v = acc[i][j][r] + bias[n];
                if (EPI == 0) {
                    ((float*)Cout)[(size_t)m * N + n] = v;
                } else {
                    ((bf16*)Cout)[(size_t)m * N + n] = (bf16)(v / (1.0f + __expf(-v)));
                }
            }
}

// ---------------------------------------------------------------------------
// Block reduction helper (256 threads = 8 waves of 32)
// ---------------------------------------------------------------------------
static __device__ inline float block_reduce_sum(float v, float* red) {
    int tid = threadIdx.x;
#pragma unroll
    for (int m = 16; m >= 1; m >>= 1) v += __shfl_xor(v, m, 32);
    if ((tid & 31) == 0) red[tid >> 5] = v;
    __syncthreads();
    float t = (tid < 8) ? red[tid] : 0.0f;
    if (tid < 32) {
#pragma unroll
        for (int m = 4; m >= 1; m >>= 1) t += __shfl_xor(t, m, 32);
        if (tid == 0) red[0] = t;
    }
    __syncthreads();
    float r = red[0];
    __syncthreads();
    return r;
}

// ---------------------------------------------------------------------------
// h = LayerNorm(x)*(1+mod[:,0:D]) + mod[:,D:2D]  -> bf16
// ---------------------------------------------------------------------------
__global__ __launch_bounds__(256) void ln_mod_kernel(
    const float* __restrict__ x, const float* __restrict__ mod,
    bf16* __restrict__ out)
{
    int row = blockIdx.x, tid = threadIdx.x;
    __shared__ float red[256];
    float v[3], s = 0.0f;
#pragma unroll
    for (int i = 0; i < 3; ++i) { v[i] = x[(size_t)row * Dm + tid + i * 256]; s += v[i]; }
    float mean = block_reduce_sum(s, red) * (1.0f / Dm);
    float s2 = 0.0f;
#pragma unroll
    for (int i = 0; i < 3; ++i) { float d = v[i] - mean; s2 += d * d; }
    float var  = block_reduce_sum(s2, red) * (1.0f / Dm);
    float rstd = rsqrtf(var + 1e-6f);
#pragma unroll
    for (int i = 0; i < 3; ++i) {
        int c = tid + i * 256;
        out[(size_t)row * Dm + c] = (bf16)(
            (v[i] - mean) * rstd * (1.0f + mod[(size_t)row * MODC + c]) +
            mod[(size_t)row * MODC + Dm + c]);
    }
}

// ---------------------------------------------------------------------------
// Per-token RMS factors folded into bf16 q/k:  qattn = q*fq*g_q, kattn = k*fk*g_k
// ---------------------------------------------------------------------------
__global__ __launch_bounds__(256) void qk_prep_kernel(
    const float* __restrict__ qkv, const float* __restrict__ g_q,
    const float* __restrict__ g_k, bf16* __restrict__ qattn, bf16* __restrict__ kattn)
{
    int row = blockIdx.x, tid = threadIdx.x;
    __shared__ float red[256];
    float q[3], k[3], sq = 0.0f, sk = 0.0f;
#pragma unroll
    for (int i = 0; i < 3; ++i) {
        int c = tid + i * 256;
        q[i] = qkv[(size_t)row * (3 * Dm) + c];
        k[i] = qkv[(size_t)row * (3 * Dm) + Dm + c];
        sq += q[i] * q[i]; sk += k[i] * k[i];
    }
    sq = block_reduce_sum(sq, red);
    sk = block_reduce_sum(sk, red);
    float fq = (SQRT_D * 0.125f) / fmaxf(sqrtf(sq), 1e-12f); // SCALE = 64^-0.5
    float fk = SQRT_D / fmaxf(sqrtf(sk), 1e-12f);
#pragma unroll
    for (int i = 0; i < 3; ++i) {
        int c = tid + i * 256;
        qattn[(size_t)row * Dm + c] = (bf16)(q[i] * fq * g_q[c]);
        kattn[(size_t)row * Dm + c] = (bf16)(k[i] * fk * g_k[c]);
    }
}

// ---------------------------------------------------------------------------
// Flash attention: grid (L/64, B*H); block 128 (4 waves x 16 query rows).
// Q/K tiles staged by the Tensor Data Mover; V manually transposed.
// ---------------------------------------------------------------------------
__global__ __launch_bounds__(128) void attn_kernel(
    const bf16* __restrict__ qattn, const bf16* __restrict__ kattn,
    const float* __restrict__ qkv, bf16* __restrict__ attn_out)
{
    const int tid  = threadIdx.x;
    const int lane = tid & 31;
    const int wv   = tid >> 5;
    const int ln   = lane & 15;
    const int half = lane >> 4;
    const int b    = blockIdx.y / Hn;
    const int h    = blockIdx.y % Hn;
    const int qb   = blockIdx.x;

    __shared__ __align__(16) bf16 Qs [64 * 72]; // [qrow][dh], TDM-padded rows
    __shared__ __align__(16) bf16 Ks [64 * 72]; // [krow][dh]
    __shared__ __align__(16) bf16 Vts[64 * 72]; // [dh][krow]
    __shared__ __align__(16) bf16 Ps [4 * 16 * 72];

    if (wv == 0)
        tdm_load_tile64(LDS_U32(Qs),
                        qattn + (size_t)(b * Lsq + qb * 64) * Dm + h * DHd, Dm);

    float   m_row[8], l_row[8];
    floatx8 oacc[4] = {};
#pragma unroll
    for (int r = 0; r < 8; ++r) { m_row[r] = -1e30f; l_row[r] = 0.0f; }

    for (int kb = 0; kb < Lsq / 64; ++kb) {
        if (wv == 0)
            tdm_load_tile64(LDS_U32(Ks),
                            kattn + (size_t)(b * Lsq + kb * 64) * Dm + h * DHd, Dm);
        // V tile, transposed, fp32 -> bf16
#pragma unroll 8
        for (int i = 0; i < 32; ++i) {
            int e = i * 128 + tid;
            int r = e >> 6, d = e & 63;
            int t = b * Lsq + kb * 64 + r;
            Vts[d * 72 + r] = (bf16)(qkv[(size_t)t * (3 * Dm) + 2 * Dm + h * DHd + d]);
        }
        if (wv == 0) __builtin_amdgcn_s_wait_tensorcnt(0);
        __syncthreads();

        // S = Q @ K^T : 16x64 per wave
        floatx8 s[4] = {};
#pragma unroll
        for (int ks = 0; ks < 2; ++ks) {
            bf16x16 aq = frag_a(&Qs[(wv * 16 + ln) * 72 + ks * 32 + half * 8]);
#pragma unroll
            for (int j = 0; j < 4; ++j) {
                bf16x16 bk = frag_b(&Ks[(j * 16 + ln) * 72 + ks * 32 + half * 8]);
                s[j] = WMMA_BF16(aq, bk, s[j]);
            }
        }

        // online softmax (row reductions across 16-lane halves)
#pragma unroll
        for (int r = 0; r < 8; ++r) {
            float sm = fmaxf(fmaxf(s[0][r], s[1][r]), fmaxf(s[2][r], s[3][r]));
#pragma unroll
            for (int m = 8; m >= 1; m >>= 1) sm = fmaxf(sm, __shfl_xor(sm, m, 32));
            float mnew = fmaxf(m_row[r], sm);
            float p0 = __expf(s[0][r] - mnew);
            float p1 = __expf(s[1][r] - mnew);
            float p2 = __expf(s[2][r] - mnew);
            float p3 = __expf(s[3][r] - mnew);
            float rs = p0 + p1 + p2 + p3;
#pragma unroll
            for (int m = 8; m >= 1; m >>= 1) rs += __shfl_xor(rs, m, 32);
            float sc = __expf(m_row[r] - mnew);
            l_row[r] = l_row[r] * sc + rs;
            m_row[r] = mnew;
#pragma unroll
            for (int j = 0; j < 4; ++j) oacc[j][r] *= sc;
            int pr = (wv * 16 + r + half * 8) * 72;
            Ps[pr + 0 * 16 + ln] = (bf16)p0;
            Ps[pr + 1 * 16 + ln] = (bf16)p1;
            Ps[pr + 2 * 16 + ln] = (bf16)p2;
            Ps[pr + 3 * 16 + ln] = (bf16)p3;
        }

        // O += P @ V (per-wave LDS round-trip; same-wave DS ops are in-order)
#pragma unroll
        for (int ks = 0; ks < 2; ++ks) {
            bf16x16 ap = frag_a(&Ps[(wv * 16 + ln) * 72 + ks * 32 + half * 8]);
#pragma unroll
            for (int j = 0; j < 4; ++j) {
                bf16x16 bv = frag_b(&Vts[(j * 16 + ln) * 72 + ks * 32 + half * 8]);
                oacc[j] = WMMA_BF16(ap, bv, oacc[j]);
            }
        }
        __syncthreads();
    }

#pragma unroll
    for (int j = 0; j < 4; ++j)
#pragma unroll
        for (int r = 0; r < 8; ++r) {
            int t   = b * Lsq + qb * 64 + wv * 16 + r + half * 8;
            int col = h * DHd + j * 16 + ln;
            attn_out[(size_t)t * Dm + col] = (bf16)(oacc[j][r] / l_row[r]);
        }
}

// ---------------------------------------------------------------------------
// Fused: a = rmsnorm(oproj)*g_o; xmid = x + g_att*a; h2 = LN(xmid)*(1+s_mlp)+sh_mlp
// ---------------------------------------------------------------------------
__global__ __launch_bounds__(256) void post_attn_kernel(
    const float* __restrict__ oproj, const float* __restrict__ g_o,
    const float* __restrict__ x, const float* __restrict__ mod,
    float* __restrict__ xmid, bf16* __restrict__ h2)
{
    int row = blockIdx.x, tid = threadIdx.x;
    __shared__ float red[256];
    float o[3], ssq = 0.0f;
#pragma unroll
    for (int i = 0; i < 3; ++i) { o[i] = oproj[(size_t)row * Dm + tid + i * 256]; ssq += o[i] * o[i]; }
    ssq = block_reduce_sum(ssq, red);
    float inv = SQRT_D / fmaxf(sqrtf(ssq), 1e-12f);
    float xm[3], s = 0.0f;
#pragma unroll
    for (int i = 0; i < 3; ++i) {
        int c = tid + i * 256;
        float a = o[i] * inv * g_o[c];
        xm[i] = x[(size_t)row * Dm + c] + mod[(size_t)row * MODC + 2 * Dm + c] * a;
        xmid[(size_t)row * Dm + c] = xm[i];
        s += xm[i];
    }
    float mean = block_reduce_sum(s, red) * (1.0f / Dm);
    float s2 = 0.0f;
#pragma unroll
    for (int i = 0; i < 3; ++i) { float d = xm[i] - mean; s2 += d * d; }
    float var  = block_reduce_sum(s2, red) * (1.0f / Dm);
    float rstd = rsqrtf(var + 1e-6f);
#pragma unroll
    for (int i = 0; i < 3; ++i) {
        int c = tid + i * 256;
        h2[(size_t)row * Dm + c] = (bf16)(
            (xm[i] - mean) * rstd * (1.0f + mod[(size_t)row * MODC + 3 * Dm + c]) +
            mod[(size_t)row * MODC + 4 * Dm + c]);
    }
}

// out = xmid + g_mlp * mlp
__global__ __launch_bounds__(256) void final_add_kernel(
    const float* __restrict__ xmid, const float* __restrict__ mod,
    const float* __restrict__ mlp, float* __restrict__ out)
{
    size_t idx = (size_t)blockIdx.x * 256 + threadIdx.x;
    int row = (int)(idx / Dm), col = (int)(idx % Dm);
    out[idx] = xmid[idx] + mod[(size_t)row * MODC + 5 * Dm + col] * mlp[idx];
}

// ---------------------------------------------------------------------------
// Workspace layout (bytes)
// ---------------------------------------------------------------------------
#define OFF_MOD    ((size_t)0)          // fp32 4096x4608   75,497,472
#define OFF_QKV    ((size_t)75497472)   // fp32 4096x2304   37,748,736
#define OFF_XMID   ((size_t)113246208)  // fp32 4096x768    12,582,912
#define OFF_OPROJ  ((size_t)125829120)  // fp32 4096x768    12,582,912 (reused: mlp)
#define OFF_H      ((size_t)138412032)  // bf16 4096x768     6,291,456 (reused: h2)
#define OFF_ASILU  ((size_t)144703488)  // bf16 4096x768     6,291,456
#define OFF_ATTN   ((size_t)150994944)  // bf16 4096x768     6,291,456
#define OFF_QATTN  ((size_t)157286400)  // bf16 4096x768     6,291,456
#define OFF_KATTN  ((size_t)163577856)  // bf16 4096x768     6,291,456
#define OFF_H1BF   ((size_t)169869312)  // bf16 4096x3072   25,165,824
#define OFF_WTA    ((size_t)195035136)  // bf16 4608x768     7,077,888
#define OFF_WTQ    ((size_t)202113024)  // bf16 2304x768     3,538,944
#define OFF_WTO    ((size_t)205651968)  // bf16 768x768      1,179,648
#define OFF_WT1    ((size_t)206831616)  // bf16 3072x768     4,718,592
#define OFF_WT2    ((size_t)211550208)  // bf16 768x3072     4,718,592

extern "C" void kernel_launch(void* const* d_in, const int* in_sizes, int n_in,
                              void* d_out, int out_size, void* d_ws, size_t ws_size,
                              hipStream_t stream) {
    (void)in_sizes; (void)n_in; (void)out_size; (void)ws_size;
    const float* x     = (const float*)d_in[0];
    const float* c     = (const float*)d_in[1];
    const float* W_qkv = (const float*)d_in[2];
    const float* b_qkv = (const float*)d_in[3];
    const float* W_o   = (const float*)d_in[4];
    const float* b_o   = (const float*)d_in[5];
    const float* g_o   = (const float*)d_in[6];
    const float* g_q   = (const float*)d_in[7];
    const float* g_k   = (const float*)d_in[8];
    const float* W1    = (const float*)d_in[9];
    const float* b1    = (const float*)d_in[10];
    const float* W2    = (const float*)d_in[11];
    const float* b2    = (const float*)d_in[12];
    const float* W_ada = (const float*)d_in[13];
    const float* b_ada = (const float*)d_in[14];

    char* w = (char*)d_ws;
    float* mod   = (float*)(w + OFF_MOD);
    float* qkv   = (float*)(w + OFF_QKV);
    float* xmid  = (float*)(w + OFF_XMID);
    float* oproj = (float*)(w + OFF_OPROJ);
    float* mlp   = oproj;
    bf16*  h     = (bf16*)(w + OFF_H);
    bf16*  h2    = h;
    bf16*  asilu = (bf16*)(w + OFF_ASILU);
    bf16*  attn  = (bf16*)(w + OFF_ATTN);
    bf16*  qattn = (bf16*)(w + OFF_QATTN);
    bf16*  kattn = (bf16*)(w + OFF_KATTN);
    bf16*  h1bf  = (bf16*)(w + OFF_H1BF);
    bf16*  WtA   = (bf16*)(w + OFF_WTA);
    bf16*  WtQ   = (bf16*)(w + OFF_WTQ);
    bf16*  WtO   = (bf16*)(w + OFF_WTO);
    bf16*  Wt1   = (bf16*)(w + OFF_WT1);
    bf16*  Wt2   = (bf16*)(w + OFF_WT2);
    float* out   = (float*)d_out;

    // Pre-pass: transpose+convert all weights to bf16 [N,K]; silu(c) to bf16.
    transpose_convert_kernel<<<dim3(MODC / 32, Dm / 32), 256, 0, stream>>>(W_ada, WtA, Dm, MODC);
    transpose_convert_kernel<<<dim3(3 * Dm / 32, Dm / 32), 256, 0, stream>>>(W_qkv, WtQ, Dm, 3 * Dm);
    transpose_convert_kernel<<<dim3(Dm / 32, Dm / 32), 256, 0, stream>>>(W_o, WtO, Dm, Dm);
    transpose_convert_kernel<<<dim3(FFd / 32, Dm / 32), 256, 0, stream>>>(W1, Wt1, Dm, FFd);
    transpose_convert_kernel<<<dim3(Dm / 32, FFd / 32), 256, 0, stream>>>(W2, Wt2, FFd, Dm);
    silu_bf16_kernel<<<(ROWS * Dm) / 256, 256, 0, stream>>>(c, asilu);

    // 1. mod = silu(c) @ W_ada + b_ada     [4096 x 4608]
    gemm_bf16_kernel<0><<<dim3(MODC / 128, ROWS / 128), 256, 0, stream>>>(
        asilu, WtA, b_ada, mod, ROWS, MODC, Dm);
    // 2. h = LN(x)*(1+s_att)+sh_att  (bf16)
    ln_mod_kernel<<<ROWS, 256, 0, stream>>>(x, mod, h);
    // 3. qkv = h @ W_qkv + b_qkv           [4096 x 2304]
    gemm_bf16_kernel<0><<<dim3(3 * Dm / 128, ROWS / 128), 256, 0, stream>>>(
        h, WtQ, b_qkv, qkv, ROWS, 3 * Dm, Dm);
    // 4. q/k RMS factors folded into bf16 qattn/kattn
    qk_prep_kernel<<<ROWS, 256, 0, stream>>>(qkv, g_q, g_k, qattn, kattn);
    // 5. flash attention (TDM-staged Q/K)
    attn_kernel<<<dim3(Lsq / 64, Bsz * Hn), 128, 0, stream>>>(qattn, kattn, qkv, attn);
    // 6. oproj = attn @ W_o + b_o          [4096 x 768]
    gemm_bf16_kernel<0><<<dim3(Dm / 128, ROWS / 128), 256, 0, stream>>>(
        attn, WtO, b_o, oproj, ROWS, Dm, Dm);
    // 7. xmid = x + g_att*rmsnorm(oproj)*g_o; h2 = LN(xmid)*(1+s_mlp)+sh_mlp
    post_attn_kernel<<<ROWS, 256, 0, stream>>>(oproj, g_o, x, mod, xmid, h2);
    // 8. h1bf = silu(h2 @ W1 + b1)  (bf16 epilogue) [4096 x 3072]
    gemm_bf16_kernel<1><<<dim3(FFd / 128, ROWS / 128), 256, 0, stream>>>(
        h2, Wt1, b1, h1bf, ROWS, FFd, Dm);
    // 9. mlp = h1bf @ W2 + b2              [4096 x 768], K = 3072
    gemm_bf16_kernel<0><<<dim3(Dm / 128, ROWS / 128), 256, 0, stream>>>(
        h1bf, Wt2, b2, mlp, ROWS, Dm, FFd);
    // 10. out = xmid + g_mlp * mlp
    final_add_kernel<<<(ROWS * Dm) / 256, 256, 0, stream>>>(xmid, mod, mlp, out);
}
